// TABlock_87067577025225
// MI455X (gfx1250) — matrix operand order repaired and checked by reading, not compiled, not run on hardware
//
#include <hip/hip_runtime.h>
#include <hip/hip_bf16.h>

// ---------------------------------------------------------------------------
// TABlock forward for MI455X (gfx1250, wave32, WMMA + TDM).
// GEMMs: bf16 in, fp32 accumulate, v_wmma_f32_16x16x32_bf16, LDS-staged via
// tensor_load_to_lds (TDM), double buffered, s_wait_tensorcnt synchronized.
// ---------------------------------------------------------------------------

typedef __attribute__((ext_vector_type(16))) __bf16 v16bf;
typedef __attribute__((ext_vector_type(8)))  float  v8f;
typedef __attribute__((ext_vector_type(4)))  unsigned int v4u;
typedef __attribute__((ext_vector_type(8)))  int    v8i;
typedef __attribute__((ext_vector_type(4)))  int    v4i;

union FragU { v16bf v; uint4 u4[2]; };

__device__ inline v8f wmma_bf16(v16bf a, v16bf b, v8f c) {
  return __builtin_amdgcn_wmma_f32_16x16x32_bf16(false, a, false, b, (short)0, c,
                                                 false, false);
}

__device__ inline float gelu_exact(float t) {
  return 0.5f * t * (1.f + erff(t * 0.70710678118654752f));
}

__device__ inline unsigned lds_byte_off(const void* p) {
  // LDS aperture: flat addr low 32 bits are the LDS byte offset.
  return (unsigned)(unsigned long long)p;
}

// ---------------------------------------------------------------------------
// TDM 2-D tile load: tile_rows x tile_k bf16 elements, row stride = ld elems,
// padded in LDS: every 16 DWORDs (32 elems) insert 4 DWORDs (8 elems)
// -> LDS row pitch 40 elements (80 B).
// ---------------------------------------------------------------------------
__device__ inline void tdm_load_2d(unsigned lds_off, const void* gptr,
                                   unsigned tile_k, unsigned tile_rows,
                                   unsigned ld) {
  const unsigned long long ga = (unsigned long long)gptr;
  v4u g0;
  g0[0] = 1u;                                   // count=1, user-mode D#
  g0[1] = lds_off;                              // lds_addr (bytes)
  g0[2] = (unsigned)(ga & 0xffffffffu);         // global_addr[31:0]
  g0[3] = (unsigned)((ga >> 32) & 0x01ffffffu)  // global_addr[56:32]
        | (2u << 30);                           // type=2 ("image")
  v8i g1;
  // dw0: wg_mask=0 | data_size=1 (2B) @16 | pad_enable @20 |
  //      pad_interval=3 (16 DW) @22 | pad_amount=3 (4 DW) @25
  g1[0] = (int)((1u << 16) | (1u << 20) | (3u << 22) | (3u << 25));
  // tensor_dim0 @ bits79:48, tensor_dim1 @ bits111:80, tile_dim0 @ bits127:112
  g1[1] = (int)((tile_k & 0xffffu) << 16);                       // td0 lo16
  g1[2] = (int)(((tile_k >> 16) & 0xffffu) |
                ((tile_rows & 0xffffu) << 16));                  // td0 hi, td1 lo
  g1[3] = (int)(((tile_rows >> 16) & 0xffffu) |
                ((tile_k & 0xffffu) << 16));                     // td1 hi, tile0
  g1[4] = (int)(tile_rows & 0xffffu);                            // tile1, tile2=0
  g1[5] = (int)ld;                                               // dim0_stride lo32
  g1[6] = 0;                                                     // stride hi, d1s lo
  g1[7] = 0;
  const v4i z4 = {0, 0, 0, 0};
  const v8i z8 = {0, 0, 0, 0, 0, 0, 0, 0};
  // 6-arg variant (clang-23 / therock headers): (g0, g1, g2, g3, extra, cpol)
  __builtin_amdgcn_tensor_load_to_lds(g0, g1, z4, z4, z8, 0);
}

// ---------------------------------------------------------------------------
// LDS-staged WMMA GEMM:  D[m][n] = sum_k A[m][k] * W[n][k]
// Block = 128x128 tile, 8 waves, each wave 32x64 (2x4 tiles of 16x16).
// K-loop step 32, TDM double buffering.
// Epilogues:
//   0: outF = acc*scale
//   1: outF = acc+bias ; outB = bf16(same)
//   2: outB = bf16(acc+bias)
//   3: outB = bf16(gelu(acc+bias))
//   4: outF = acc+bias+res
//   5: outT[b*sTb + n*ldT + (m%rowsPB)] = bf16(acc+bias)
// ---------------------------------------------------------------------------
#define LPITCH 40  // padded LDS row pitch in bf16 elements (80 B)

template <int EPI>
__global__ __launch_bounds__(256) void gemm_wmma_kernel(
    const __bf16* __restrict__ A, int lda, long long sAb,
    const __bf16* __restrict__ W, int ldw, long long sWb,
    const float* __restrict__ bias, float scale,
    float* __restrict__ outF, __bf16* __restrict__ outB, int ldo, long long sOb,
    const float* __restrict__ res,
    __bf16* __restrict__ outT, int ldT, int rowsPB, long long sTb,
    int M, int Nout, int K) {
  __shared__ __bf16 As[2][128 * LPITCH];
  __shared__ __bf16 Bs[2][128 * LPITCH];

  const int wave = threadIdx.x >> 5;
  const int lane = threadIdx.x & 31;
  const int MB = M >> 7;
  const int mb0 = (blockIdx.x % MB) << 7;
  const int nb0 = (blockIdx.x / MB) << 7;

  const long long bA = (long long)blockIdx.y * sAb;
  const long long bW = (long long)blockIdx.y * sWb;
  const long long bO = (long long)blockIdx.y * sOb;

  const int rw = wave & 3;   // 4 row groups of 32
  const int cw = wave >> 2;  // 2 col groups of 64
  const int half = lane >> 4;
  const int l15 = lane & 15;
  const int kbA = half << 3;   // A frag K-base: 0 or 8 (+16 second block)
  const int kbB = half << 3;   // B frag K-base: 0 or 8 (+16 second block)

  const v8f zero = {0.f, 0.f, 0.f, 0.f, 0.f, 0.f, 0.f, 0.f};
  v8f acc[2][4];
#pragma unroll
  for (int mi = 0; mi < 2; ++mi)
#pragma unroll
    for (int j = 0; j < 4; ++j) acc[mi][j] = zero;

  const __bf16* Ag = A + bA + (long long)mb0 * lda;
  const __bf16* Wg = W + bW + (long long)nb0 * ldw;
  const int KT = K >> 5;
  const bool issuer = (wave == 0);

  if (issuer) {
    tdm_load_2d(lds_byte_off(&As[0][0]), Ag, 32, 128, (unsigned)lda);
    tdm_load_2d(lds_byte_off(&Bs[0][0]), Wg, 32, 128, (unsigned)ldw);
    __builtin_amdgcn_s_wait_tensorcnt(0);
  }
  __syncthreads();

  for (int kt = 0; kt < KT; ++kt) {
    const int buf = kt & 1;
    if (issuer && (kt + 1) < KT) {  // prefetch next stage into other buffer
      tdm_load_2d(lds_byte_off(&As[buf ^ 1][0]), Ag + ((kt + 1) << 5), 32, 128,
                  (unsigned)lda);
      tdm_load_2d(lds_byte_off(&Bs[buf ^ 1][0]), Wg + ((kt + 1) << 5), 32, 128,
                  (unsigned)ldw);
    }

    FragU afr[2];
#pragma unroll
    for (int mi = 0; mi < 2; ++mi) {
      const __bf16* ap = &As[buf][(rw * 32 + mi * 16 + l15) * LPITCH + kbA];
      afr[mi].u4[0] = *(const uint4*)ap;         // K = kbA..kbA+7
      afr[mi].u4[1] = *(const uint4*)(ap + 16);  // K = kbA+16..kbA+23
    }
#pragma unroll
    for (int j = 0; j < 4; ++j) {
      const __bf16* bp = &Bs[buf][(cw * 64 + j * 16 + l15) * LPITCH + kbB];
      FragU bfr;
      bfr.u4[0] = *(const uint4*)bp;        // K = kbB..kbB+7
      bfr.u4[1] = *(const uint4*)(bp + 16);  // K = kbB+16..kbB+23
      acc[0][j] = wmma_bf16(afr[0].v, bfr.v, acc[0][j]);
      acc[1][j] = wmma_bf16(afr[1].v, bfr.v, acc[1][j]);
    }

    if (issuer) __builtin_amdgcn_s_wait_tensorcnt(0);  // next stage landed
    __syncthreads();
  }

  // Epilogue. C/D layout: lanes 0-15 -> rows +0..7, lanes 16-31 -> rows +8..15.
#pragma unroll
  for (int mi = 0; mi < 2; ++mi) {
    const int mbase = mb0 + rw * 32 + mi * 16 + (half << 3);
#pragma unroll
    for (int j = 0; j < 4; ++j) {
      const int n = nb0 + cw * 64 + j * 16 + l15;
      float bi = 0.f;
      if (EPI != 0) bi = bias[n];
#pragma unroll
      for (int r = 0; r < 8; ++r) {
        const float v = acc[mi][j][r];
        const int m = mbase + r;
        const long long o = bO + (long long)m * ldo + n;
        if (EPI == 0) {
          outF[o] = v * scale;
        } else if (EPI == 1) {
          const float s = v + bi;
          outF[o] = s;
          outB[o] = (__bf16)s;
        } else if (EPI == 2) {
          outB[o] = (__bf16)(v + bi);
        } else if (EPI == 3) {
          outB[o] = (__bf16)gelu_exact(v + bi);
        } else if (EPI == 4) {
          outF[o] = v + bi + res[o];
        } else if (EPI == 5) {
          const int bb = m / rowsPB;
          const int mr = m - bb * rowsPB;
          outT[(long long)bb * sTb + (long long)n * ldT + mr] = (__bf16)(v + bi);
        }
      }
    }
  }
}

// ---------------------------------------------------------------------------
// f32 -> bf16 convert
// ---------------------------------------------------------------------------
__global__ void f32_to_bf16_kernel(const float* __restrict__ in,
                                   __bf16* __restrict__ out, long long n) {
  long long i = (long long)blockIdx.x * blockDim.x + threadIdx.x;
  if (i < n) out[i] = (__bf16)in[i];
}

// ---------------------------------------------------------------------------
// Depthwise 7x7 conv on q (stored (b, n, c) f32), groups G=256, pad 3.
// ---------------------------------------------------------------------------
__global__ __launch_bounds__(256) void dwconv_kernel(
    const float* __restrict__ q, const float* __restrict__ dww,
    const float* __restrict__ dwb, float* __restrict__ off) {
  const long long idx = (long long)blockIdx.x * 256 + threadIdx.x;
  const int g = (int)(idx & 255);
  const int n = (int)((idx >> 8) & 1023);
  const int b3 = (int)(idx >> 18);
  if (b3 >= 24) return;
  const int h = n >> 5, w = n & 31;
  const int b = b3 / 3, cbase = (b3 % 3) * 256;
  const float* wk = dww + g * 49;
  float acc = dwb[g];
#pragma unroll
  for (int dy = 0; dy < 7; ++dy) {
    const int yy = h + dy - 3;
    if (yy < 0 || yy > 31) continue;
#pragma unroll
    for (int dx = 0; dx < 7; ++dx) {
      const int xx = w + dx - 3;
      if (xx < 0 || xx > 31) continue;
      acc += wk[dy * 7 + dx] *
             q[((long long)b * 1024 + yy * 32 + xx) * 768 + cbase + g];
    }
  }
  off[((long long)b3 * 1024 + n) * 256 + g] = acc;
}

// ---------------------------------------------------------------------------
// Fused LN(G=256) + GELU + pointwise(2xG) + tanh + grid build.
// ---------------------------------------------------------------------------
__global__ __launch_bounds__(256) void off_ln_pw_kernel(
    const float* __restrict__ off, const float* __restrict__ lnw,
    const float* __restrict__ lnb, const float* __restrict__ pww,
    float* __restrict__ grid) {
  __shared__ float red[256];
  const long long bn = blockIdx.x;
  const int g = threadIdx.x;
  const float v = off[bn * 256 + g];

  red[g] = v; __syncthreads();
  for (int s = 128; s > 0; s >>= 1) { if (g < s) red[g] += red[g + s]; __syncthreads(); }
  const float mu = red[0] * (1.f / 256.f); __syncthreads();

  const float d = v - mu;
  red[g] = d * d; __syncthreads();
  for (int s = 128; s > 0; s >>= 1) { if (g < s) red[g] += red[g + s]; __syncthreads(); }
  const float inv = rsqrtf(red[0] * (1.f / 256.f) + 1e-5f); __syncthreads();

  float a = d * inv * lnw[g] + lnb[g];
  a = gelu_exact(a);

  red[g] = a * pww[g]; __syncthreads();
  for (int s = 128; s > 0; s >>= 1) { if (g < s) red[g] += red[g + s]; __syncthreads(); }
  const float t0 = red[0]; __syncthreads();

  red[g] = a * pww[256 + g]; __syncthreads();
  for (int s = 128; s > 0; s >>= 1) { if (g < s) red[g] += red[g + s]; __syncthreads(); }
  const float t1 = red[0];

  if (g == 0) {
    const int n = (int)(bn & 1023);
    const int h = n >> 5, w = n & 31;
    const float ry = ((h + 0.5f) / 32.f) * 2.f - 1.f;
    const float rx = ((w + 0.5f) / 32.f) * 2.f - 1.f;
    grid[bn * 2 + 0] = tanhf(t1) * (2.f / 32.f) + rx;  // gx
    grid[bn * 2 + 1] = tanhf(t0) * (2.f / 32.f) + ry;  // gy
  }
}

// ---------------------------------------------------------------------------
// Bilinear grid sample from x (B,N,C) channel view -> xs bf16 (b, n, c)
// ---------------------------------------------------------------------------
__global__ __launch_bounds__(256) void sample_kernel(
    const float* __restrict__ x, const float* __restrict__ grid,
    __bf16* __restrict__ xs) {
  const long long bn = blockIdx.x;
  const int g = threadIdx.x;
  const int b3 = (int)(bn >> 10);
  const int n = (int)(bn & 1023);
  const int b = b3 / 3, cbase = (b3 % 3) * 256;
  const float gx = grid[bn * 2 + 0];
  const float gy = grid[bn * 2 + 1];
  const float fx = (gx + 1.f) * 0.5f * 31.f;
  const float fy = (gy + 1.f) * 0.5f * 31.f;
  const float x0 = floorf(fx), y0 = floorf(fy);
  const float wx = fx - x0, wy = fy - y0;
  const float xi[4] = {x0, x0 + 1.f, x0, x0 + 1.f};
  const float yi[4] = {y0, y0, y0 + 1.f, y0 + 1.f};
  const float wt[4] = {(1.f - wx) * (1.f - wy), wx * (1.f - wy),
                       (1.f - wx) * wy, wx * wy};
  float acc = 0.f;
#pragma unroll
  for (int i = 0; i < 4; ++i) {
    const bool valid =
        xi[i] >= 0.f && xi[i] <= 31.f && yi[i] >= 0.f && yi[i] <= 31.f;
    if (valid) {
      const int xc = (int)xi[i], yc = (int)yi[i];
      acc += wt[i] * x[((long long)b * 1024 + yc * 32 + xc) * 768 + cbase + g];
    }
  }
  xs[((long long)b * 1024 + n) * 768 + cbase + g] = (__bf16)acc;
}

// ---------------------------------------------------------------------------
// Row softmax over 1024, write bf16 probs.
// ---------------------------------------------------------------------------
__global__ __launch_bounds__(256) void softmax_kernel(
    const float* __restrict__ S, __bf16* __restrict__ P) {
  __shared__ float red[256];
  const long long row = blockIdx.x;
  const float* s = S + row * 1024;
  const int t = threadIdx.x;
  float v0 = s[t], v1 = s[t + 256], v2 = s[t + 512], v3 = s[t + 768];
  float mx = fmaxf(fmaxf(v0, v1), fmaxf(v2, v3));
  red[t] = mx; __syncthreads();
  for (int st = 128; st > 0; st >>= 1) { if (t < st) red[t] = fmaxf(red[t], red[t + st]); __syncthreads(); }
  const float M = red[0]; __syncthreads();
  v0 = expf(v0 - M); v1 = expf(v1 - M); v2 = expf(v2 - M); v3 = expf(v3 - M);
  red[t] = v0 + v1 + v2 + v3; __syncthreads();
  for (int st = 128; st > 0; st >>= 1) { if (t < st) red[t] += red[t + st]; __syncthreads(); }
  const float inv = 1.f / red[0];
  __bf16* p = P + row * 1024;
  p[t]       = (__bf16)(v0 * inv);
  p[t + 256] = (__bf16)(v1 * inv);
  p[t + 512] = (__bf16)(v2 * inv);
  p[t + 768] = (__bf16)(v3 * inv);
}

// ---------------------------------------------------------------------------
// LayerNorm over C=768, write bf16.
// ---------------------------------------------------------------------------
__global__ __launch_bounds__(256) void ln768_kernel(
    const float* __restrict__ in, const float* __restrict__ w,
    const float* __restrict__ b, __bf16* __restrict__ out) {
  __shared__ float red[256];
  const long long row = blockIdx.x;
  const float* r = in + row * 768;
  const int t = threadIdx.x;
  const float v0 = r[t], v1 = r[t + 256], v2 = r[t + 512];
  red[t] = v0 + v1 + v2; __syncthreads();
  for (int s = 128; s > 0; s >>= 1) { if (t < s) red[t] += red[t + s]; __syncthreads(); }
  const float mu = red[0] * (1.f / 768.f); __syncthreads();
  const float d0 = v0 - mu, d1 = v1 - mu, d2 = v2 - mu;
  red[t] = d0 * d0 + d1 * d1 + d2 * d2; __syncthreads();
  for (int s = 128; s > 0; s >>= 1) { if (t < s) red[t] += red[t + s]; __syncthreads(); }
  const float inv = rsqrtf(red[0] * (1.f / 768.f) + 1e-5f);
  __bf16* o = out + row * 768;
  o[t]       = (__bf16)(d0 * inv * w[t] + b[t]);
  o[t + 256] = (__bf16)(d1 * inv * w[t + 256] + b[t + 256]);
  o[t + 512] = (__bf16)(d2 * inv * w[t + 512] + b[t + 512]);
}

// ---------------------------------------------------------------------------
// Host orchestration
// ---------------------------------------------------------------------------
extern "C" void kernel_launch(void* const* d_in, const int* in_sizes, int n_in,
                              void* d_out, int out_size, void* d_ws, size_t ws_size,
                              hipStream_t stream) {
  (void)in_sizes; (void)n_in; (void)out_size; (void)ws_size;
  const float* x    = (const float*)d_in[0];
  const float* Wq   = (const float*)d_in[1];
  const float* bq   = (const float*)d_in[2];
  const float* Wk   = (const float*)d_in[3];
  const float* bk   = (const float*)d_in[4];
  const float* Wv   = (const float*)d_in[5];
  const float* bv   = (const float*)d_in[6];
  const float* dw_w = (const float*)d_in[7];
  const float* dw_b = (const float*)d_in[8];
  const float* lnow = (const float*)d_in[9];
  const float* lnob = (const float*)d_in[10];
  const float* pw_w = (const float*)d_in[11];
  const float* ln1w = (const float*)d_in[12];
  const float* ln1b = (const float*)d_in[13];
  const float* ln2w = (const float*)d_in[14];
  const float* ln2b = (const float*)d_in[15];
  const float* fc1w = (const float*)d_in[16];
  const float* fc1b = (const float*)d_in[17];
  const float* fc2w = (const float*)d_in[18];
  const float* fc2b = (const float*)d_in[19];
  float* out = (float*)d_out;

  const int Bn = 8, N = 1024, C = 768, Hd = 3072;
  const long long MN = (long long)Bn * N;  // 8192

  size_t wsoff = 0;
  auto alloc = [&](size_t bytes) -> void* {
    void* p = (char*)d_ws + wsoff;
    wsoff += (bytes + 255) & ~(size_t)255;
    return p;
  };

  __bf16* x_bf   = (__bf16*)alloc(MN * C * 2);
  __bf16* wq_bf  = (__bf16*)alloc((size_t)C * C * 2);
  __bf16* wk_bf  = (__bf16*)alloc((size_t)C * C * 2);
  __bf16* wv_bf  = (__bf16*)alloc((size_t)C * C * 2);
  __bf16* f1_bf  = (__bf16*)alloc((size_t)Hd * C * 2);
  __bf16* f2_bf  = (__bf16*)alloc((size_t)C * Hd * 2);
  float*  q_f    = (float*)alloc(MN * C * 4);
  __bf16* q_bf   = (__bf16*)alloc(MN * C * 2);
  float*  off_f  = (float*)alloc((size_t)24 * 1024 * 256 * 4);
  float*  grid_f = (float*)alloc((size_t)24 * 1024 * 2 * 4);
  __bf16* xs_bf  = (__bf16*)alloc(MN * C * 2);
  __bf16* k_bf   = (__bf16*)alloc(MN * C * 2);
  __bf16* vT_bf  = (__bf16*)alloc(MN * C * 2);  // (b, c, n)
  float*  S_f    = (float*)alloc((size_t)Bn * N * N * 4);
  __bf16* P_bf   = (__bf16*)alloc((size_t)Bn * N * N * 2);
  float*  ao_f   = (float*)alloc(MN * C * 4);   // attention out (b,n,c)
  __bf16* ln_bf  = (__bf16*)alloc(MN * C * 2);
  __bf16* h_bf   = (__bf16*)alloc(MN * Hd * 2);
  float*  x1_f   = (float*)alloc(MN * C * 4);

  auto cvt = [&](const float* src, __bf16* dst, long long n) {
    f32_to_bf16_kernel<<<(unsigned)((n + 255) / 256), 256, 0, stream>>>(src, dst, n);
  };
  cvt(x, x_bf, MN * C);
  cvt(Wq, wq_bf, (long long)C * C);
  cvt(Wk, wk_bf, (long long)C * C);
  cvt(Wv, wv_bf, (long long)C * C);
  cvt(fc1w, f1_bf, (long long)Hd * C);
  cvt(fc2w, f2_bf, (long long)C * Hd);

  auto nblk = [](int M, int Nout) { return (unsigned)((M >> 7) * (Nout >> 7)); };

  // ---- Q projection: q = x @ Wq^T + bq -> q_f (f32) + q_bf
  gemm_wmma_kernel<1><<<dim3(nblk((int)MN, C), 1), 256, 0, stream>>>(
      x_bf, C, 0, wq_bf, C, 0, bq, 1.f, q_f, q_bf, C, 0, nullptr,
      nullptr, 0, 0, 0, (int)MN, C, C);

  // ---- Offset path
  dwconv_kernel<<<24576, 256, 0, stream>>>(q_f, dw_w, dw_b, off_f);
  off_ln_pw_kernel<<<24576, 256, 0, stream>>>(off_f, lnow, lnob, pw_w, grid_f);
  sample_kernel<<<24576, 256, 0, stream>>>(x, grid_f, xs_bf);

  // ---- K projection
  gemm_wmma_kernel<2><<<dim3(nblk((int)MN, C), 1), 256, 0, stream>>>(
      xs_bf, C, 0, wk_bf, C, 0, bk, 1.f, nullptr, k_bf, C, 0, nullptr,
      nullptr, 0, 0, 0, (int)MN, C, C);

  // ---- V projection (transposed store -> (b, c, n))
  gemm_wmma_kernel<5><<<dim3(nblk((int)MN, C), 1), 256, 0, stream>>>(
      xs_bf, C, 0, wv_bf, C, 0, bv, 1.f, nullptr, nullptr, 0, 0, nullptr,
      vT_bf, N, N, (long long)C * N, (int)MN, C, C);

  // ---- Attention scores (batched over blockIdx.y)
  const float scale = rsqrtf((float)C);
  gemm_wmma_kernel<0><<<dim3(nblk(N, N), Bn), 256, 0, stream>>>(
      q_bf, C, (long long)N * C, k_bf, C, (long long)N * C, nullptr, scale,
      S_f, nullptr, N, (long long)N * N, nullptr, nullptr, 0, 0, 0, N, N, C);

  // ---- Softmax
  softmax_kernel<<<(unsigned)MN, 256, 0, stream>>>(S_f, P_bf);

  // ---- out = P @ v  (W = vT row-major (c, n))
  gemm_wmma_kernel<0><<<dim3(nblk(N, C), Bn), 256, 0, stream>>>(
      P_bf, N, (long long)N * N, vT_bf, N, (long long)C * N, nullptr, 1.f,
      ao_f, nullptr, C, (long long)N * C, nullptr, nullptr, 0, 0, 0, N, C, N);

  // ---- MLP 1: x1 = x + fc2(gelu(fc1(LN1(out))))
  ln768_kernel<<<(unsigned)MN, 256, 0, stream>>>(ao_f, ln1w, ln1b, ln_bf);
  gemm_wmma_kernel<3><<<dim3(nblk((int)MN, Hd), 1), 256, 0, stream>>>(
      ln_bf, C, 0, f1_bf, C, 0, fc1b, 1.f, nullptr, h_bf, Hd, 0, nullptr,
      nullptr, 0, 0, 0, (int)MN, Hd, C);
  gemm_wmma_kernel<4><<<dim3(nblk((int)MN, C), 1), 256, 0, stream>>>(
      h_bf, Hd, 0, f2_bf, Hd, 0, fc2b, 1.f, x1_f, nullptr, C, 0, x,
      nullptr, 0, 0, 0, (int)MN, C, Hd);

  // ---- MLP 2: x2 = x1 + fc2(gelu(fc1(LN2(x1)))) -> d_out
  ln768_kernel<<<(unsigned)MN, 256, 0, stream>>>(x1_f, ln2w, ln2b, ln_bf);
  gemm_wmma_kernel<3><<<dim3(nblk((int)MN, Hd), 1), 256, 0, stream>>>(
      ln_bf, C, 0, f1_bf, C, 0, fc1b, 1.f, nullptr, h_bf, Hd, 0, nullptr,
      nullptr, 0, 0, 0, (int)MN, Hd, C);
  gemm_wmma_kernel<4><<<dim3(nblk((int)MN, C), 1), 256, 0, stream>>>(
      h_bf, Hd, 0, f2_bf, Hd, 0, fc2b, 1.f, out, nullptr, C, 0, x1_f,
      nullptr, 0, 0, 0, (int)MN, C, Hd);
}
